// LogisticTensorFactorModel_23072564314606
// MI455X (gfx1250) — matrix operand order repaired and checked by reading, not compiled, not run on hardware
//
#include <hip/hip_runtime.h>

typedef __attribute__((ext_vector_type(16))) _Float16 v16h;
typedef __attribute__((ext_vector_type(8)))  float    v8f;
typedef __attribute__((ext_vector_type(4)))  float    v4f;

#define I_DIM 1024
#define J_DIM 512
#define K_DIM 256
#define D_DIM 64

// CP reconstruction + sigmoid:
//   out[i,j,k] = sigmoid( sum_d W[i,d] * V[j,d] * U[k,d] )
// GEMM view per j: C[16i x 16k] = A(Wtile, 16x64 f16) x B(64x16 f16), B[d,k]=V[j,d]*U[k,d]
// One wave owns one (i-tile, k-tile) pair and loops over a strip of j values,
// issuing two v_wmma_f32_16x16x32_f16 per j (d split 0..31 / 32..63).
__global__ __launch_bounds__(256) void LogisticCP_wmma_kernel(
    const float* __restrict__ W, const float* __restrict__ V,
    const float* __restrict__ U, float* __restrict__ out)
{
    const int lane = threadIdx.x & 31;
    const int wave = threadIdx.x >> 5;   // 0..7
    const int m    = lane & 15;          // row (A/U) and column n (B/C/D)
    const int half = lane >> 4;          // lane-half selects d-subrange

    const int pair  = blockIdx.x;        // 0..1023
    const int iBase = (pair >> 4) * 16;  // 64 i-tiles
    const int kBase = (pair & 15) * 16;  // 16 k-tiles

    // ---------- A fragments: W[iBase+m, d] in contiguous k order, same as B ----------
    // lane (m, half), element e of a0 <-> d = 16*half + e
    // a1 is the same with d += 32.
    const float* wrow = W + (iBase + m) * D_DIM + 16 * half;
    float wf[32];
    #pragma unroll
    for (int c = 0; c < 4; ++c) {
        v4f lo = *(const v4f*)(wrow + 8 * c + 16 * (c >> 1));
        v4f hi4 = *(const v4f*)(wrow + 8 * c + 16 * (c >> 1) + 4);
        #pragma unroll
        for (int t = 0; t < 4; ++t) { wf[8 * c + t] = lo[t]; wf[8 * c + 4 + t] = hi4[t]; }
    }
    v16h a0, a1;
    #pragma unroll
    for (int e = 0; e < 16; ++e) {
        a0[e] = (_Float16)wf[e];
        a1[e] = (_Float16)wf[16 + e];
    }

    // ---------- U fragments (B-layout factor): element e <-> d = 16*half + e ----------
    const float* urow = U + (kBase + m) * D_DIM + 16 * half;
    v16h u0, u1;
    #pragma unroll
    for (int c = 0; c < 4; ++c) {
        v4f q0 = *(const v4f*)(urow + 4 * c);        // d in [16*half, 16*half+16)
        v4f q1 = *(const v4f*)(urow + 32 + 4 * c);   // d in [32+16*half, ...)
        #pragma unroll
        for (int t = 0; t < 4; ++t) {
            u0[4 * c + t] = (_Float16)q0[t];
            u1[4 * c + t] = (_Float16)q1[t];
        }
    }

    // ---------- j loop: rebuild B = (V row) * (U frag), 2 WMMAs, sigmoid, store ----------
    const int jStart = blockIdx.y * 8 + wave;        // gridDim.y = 8, 8 waves -> stride 64
    for (int j = jStart; j < J_DIM; j += 64) {
        const float* vbase = V + j * D_DIM + 16 * half;   // wave-uniform row, L2-resident
        v4f vv[8];
        #pragma unroll
        for (int c = 0; c < 4; ++c) {
            vv[c]     = *(const v4f*)(vbase + 4 * c);
            vv[4 + c] = *(const v4f*)(vbase + 32 + 4 * c);
        }
        v16h b0, b1;
        #pragma unroll
        for (int e = 0; e < 16; ++e) {
            b0[e] = (_Float16)(vv[e >> 2][e & 3])       * u0[e];  // v_pk_mul_f16
            b1[e] = (_Float16)(vv[4 + (e >> 2)][e & 3]) * u1[e];
        }

        v8f acc = {};
        acc = __builtin_amdgcn_wmma_f32_16x16x32_f16(false, a0, false, b0,
                                                     (short)0, acc, false, false);
        acc = __builtin_amdgcn_wmma_f32_16x16x32_f16(false, a1, false, b1,
                                                     (short)0, acc, false, false);

        // C/D layout: VGPR r, this lane -> out row i = iBase + r + 8*half, col k = kBase + m
        #pragma unroll
        for (int r = 0; r < 8; ++r) {
            float x = acc[r];
            // sigmoid(x) = 1 / (1 + exp2(-x * log2(e)))
            float e2 = __builtin_amdgcn_exp2f(-x * 1.442695040888963f);
            float s  = __builtin_amdgcn_rcpf(1.0f + e2);
            size_t idx = (((size_t)(iBase + r + 8 * half)) * J_DIM + j) * K_DIM
                         + (kBase + m);
            out[idx] = s;
        }
    }
}

extern "C" void kernel_launch(void* const* d_in, const int* in_sizes, int n_in,
                              void* d_out, int out_size, void* d_ws, size_t ws_size,
                              hipStream_t stream) {
    const float* W = (const float*)d_in[0];
    const float* V = (const float*)d_in[1];
    const float* U = (const float*)d_in[2];
    float* out = (float*)d_out;

    dim3 grid(1024, 8);   // 1024 (i-tile,k-tile) pairs x 8 j-strips
    dim3 block(256);      // 8 wave32 waves; each wave owns one tile pair, 8 j's
    LogisticCP_wmma_kernel<<<grid, block, 0, stream>>>(W, V, U, out);
}
